// EdgeEmbedding_57526791962766
// MI455X (gfx1250) — matrix operand order repaired and checked by reading, not compiled, not run on hardware
//
#include <hip/hip_runtime.h>
#include <math.h>

typedef __attribute__((ext_vector_type(16))) _Float16 v16h;
typedef __attribute__((ext_vector_type(8)))  float    v8f;

#define HIDDEN 128
#define NPTS   512
#define BATCH  2

// ---------------------------------------------------------------------------
// Pass 1: permute W (128x129 f32) into f16 WMMA B-fragment order in d_ws.
// Layout: [kc(4)][nt(8)][lane(32)][16 halves]  = 16384 halves = 32 KB.
// B-matrix 32x16 f16 layout used here (contiguous K, not ISA table order): for column n = lane&15, the 16
// halves of a lane are K = kk..kk+15 with kk = (lane>>4)*16 within the chunk.
// ---------------------------------------------------------------------------
__global__ __launch_bounds__(256)
void PermuteW_kernel(const float* __restrict__ W, _Float16* __restrict__ wsW)
{
    const int t = blockIdx.x * 256 + threadIdx.x;   // 0..16383
    const int kc   = t >> 12;
    const int nt   = (t >> 9) & 7;
    const int lane = (t >> 4) & 31;
    const int j    = t & 15;
    const int n    = lane & 15;
    const int kk   = (lane >> 4) * 16;
    const int k    = kc * 32 + kk + j;              // feature 0..127 -> W col k+1
    const int h    = nt * 16 + n;                   // output column
    wsW[t] = (_Float16)W[h * (HIDDEN + 1) + (k + 1)];
}

// ---------------------------------------------------------------------------
// Pass 2: main fused kernel.
// Block = 256 threads = 8 waves; each wave computes a 16(rows=j) x 128(h)
// output tile via 8 column-tiles x 4 K-chunks of V_WMMA_F32_16X16X32_F16.
// Grid = BATCH * NPTS * (NPTS/128) = 4096 blocks.
// ---------------------------------------------------------------------------
__global__ __launch_bounds__(256)
void EdgeEmbedding_wmma_kernel(const float* __restrict__ coords, // (B,N,2)
                               const float* __restrict__ adj_t,  // (B,N,N)
                               const float* __restrict__ W,      // (128,129) f32 (col 0 + bias path)
                               const float* __restrict__ bias,   // (128,)
                               const _Float16* __restrict__ wsW, // permuted f16 B image
                               float* __restrict__ out)          // (B,N,N,128)
{
    __shared__ _Float16 ldsW[4 * 8 * 32 * 16];   // 32 KB, fragment order
    __shared__ float    ldsDist[8][16];          // per-wave row distances

    const int tid = threadIdx.x;

    // ---- stage permuted W image: straight 32 KB copy, b128 granularity ----
    {
        const int4* __restrict__ src = (const int4*)wsW;
        int4* __restrict__ dst = (int4*)ldsW;
        __builtin_prefetch(&src[tid], 0, 1);     // global_prefetch_b8
        #pragma unroll
        for (int t = tid; t < 2048; t += 256)
            dst[t] = src[t];
    }

    const int wave = tid >> 5;
    const int lane = tid & 31;
    const int m    = lane & 15;   // row within 16-row tile
    const int hi   = lane >> 4;   // half-wave select

    // ---- decode block -> (batch, i, j-block) -----------------------------
    const int blk = blockIdx.x;
    const int jb  = blk & 3;                  // which 128-wide j slab
    const int i   = (blk >> 2) & (NPTS - 1);
    const int bb  = blk >> 11;                // blk / (512*4)

    const int       jrow    = jb * 128 + wave * 16 + m;
    const long long rowbase = ((long long)(bb * NPTS + i)) * NPTS + jb * 128 + wave * 16;

    // ---- per-row scalars: adj value + pairwise distance ------------------
    const float adj = adj_t[((long long)(bb * NPTS + i)) * NPTS + jrow];
    const float cix = coords[(bb * NPTS + i) * 2 + 0];
    const float ciy = coords[(bb * NPTS + i) * 2 + 1];
    const float cjx = coords[(bb * NPTS + jrow) * 2 + 0];
    const float cjy = coords[(bb * NPTS + jrow) * 2 + 1];
    const float dx = cix - cjx, dy = ciy - cjy;
    ldsDist[wave][m] = sqrtf(dx * dx + dy * dy);  // both half-waves write same value

    // ---- build A fragments: sin/cos positional features ------------------
    // A-matrix 16x32 f16 layout (matches B image):
    //   lane<16 : halves 0..15 -> K=0..15
    //   lane>=16: halves 0..15 -> K=16..31
    const float NEG_LN1E4_OVER_64 = -0.14391156516f;  // -ln(10000)/64
    v16h afrag[4];
    #pragma unroll
    for (int kc = 0; kc < 4; ++kc) {
        #pragma unroll
        for (int jj = 0; jj < 16; ++jj) {
            const int K = hi * 16 + jj;
            const int k = kc * 32 + K;                 // global feature 0..127
            const int f = k & 63;                      // frequency index
            const float arg = adj * __expf((float)f * NEG_LN1E4_OVER_64);
            const float v   = (k < 64) ? __sinf(arg) : __cosf(arg);
            afrag[kc][jj] = (_Float16)v;
        }
    }

    __syncthreads();   // ldsW + ldsDist ready

    // ---- 8 column tiles x 4 K-chunks of WMMA -----------------------------
    const int n = lane & 15;
    #pragma unroll
    for (int nt = 0; nt < 8; ++nt) {
        v8f c = {};
        #pragma unroll
        for (int kc = 0; kc < 4; ++kc) {
            const v16h bfrag = *(const v16h*)&ldsW[kc * 4096 + nt * 512 + lane * 16];
            c = __builtin_amdgcn_wmma_f32_16x16x32_f16(
                    /*neg_a=*/false, afrag[kc],
                    /*neg_b=*/false, bfrag,
                    /*c_mod=*/(short)0, c,
                    /*reuse_a=*/false, /*reuse_b=*/false);
        }
        // epilogue: + dist*W[h][0] + bias[h], ReLU
        const int   h  = nt * 16 + n;
        const float w0 = W[h * (HIDDEN + 1)];
        const float bv = bias[h];
        #pragma unroll
        for (int r = 0; r < 8; ++r) {
            const int   rm = r + hi * 8;                 // C layout: lanes>=16 hold M=r+8
            const float d  = ldsDist[wave][rm];
            float v = c[r] + d * w0 + bv;
            v = v > 0.0f ? v : 0.0f;
            out[(rowbase + rm) * HIDDEN + h] = v;
        }
    }
}

extern "C" void kernel_launch(void* const* d_in, const int* in_sizes, int n_in,
                              void* d_out, int out_size, void* d_ws, size_t ws_size,
                              hipStream_t stream) {
    const float* coords = (const float*)d_in[0];
    const float* adj_t  = (const float*)d_in[1];
    const float* W      = (const float*)d_in[2];
    const float* b      = (const float*)d_in[3];
    float* out = (float*)d_out;
    _Float16* wsW = (_Float16*)d_ws;          // 32 KB of workspace used

    // Pass 1: build permuted f16 B image (re-run every call: deterministic,
    // workspace contents never assumed to persist).
    PermuteW_kernel<<<dim3(64), dim3(256), 0, stream>>>(W, wsW);

    // Pass 2: fused feature-gen + WMMA GEMM + epilogue.
    dim3 grid(BATCH * NPTS * (NPTS / 128));   // 4096 blocks
    dim3 block(256);                           // 8 wave32
    EdgeEmbedding_wmma_kernel<<<grid, block, 0, stream>>>(coords, adj_t, W, b, wsW, out);
}
